// TransformerBlock_45079976739090
// MI455X (gfx1250) — matrix operand order repaired and checked by reading, not compiled, not run on hardware
//
#include <hip/hip_runtime.h>
#include <hip/hip_bf16.h>

// ---------------------------------------------------------------------------
// MI455X (gfx1250) transformer block.
//  B=4, N=2048, H=16, D=64, DM=1024.
//  ~155 GFLOP total, ~350MB HBM traffic -> compute bound -> route all matmuls
//  through v_wmma_f32_16x16x32_bf16 (bf16 operands, fp32 accumulate).
//  GEMM pipeline: double-buffered LDS tiles; global->LDS via
//  GLOBAL_LOAD_ASYNC_TO_LDS_B128 (ASYNCcnt) when the builtin exists,
//  else reg-staged loads.
// ---------------------------------------------------------------------------

#define DMODEL 1024
#define NSEQ   2048
#define BATCH  4
#define NHEAD  16
#define DHEAD  64

typedef __attribute__((ext_vector_type(16))) __bf16 bf16x16;
typedef __attribute__((ext_vector_type(8)))  __bf16 bf16x8;
typedef __attribute__((ext_vector_type(8)))  float  f32x8;

#if defined(__has_builtin)
#if __has_builtin(__builtin_amdgcn_global_load_async_to_lds_b128)
#define ASYNC_LDS 1
#endif
#if __has_builtin(__builtin_amdgcn_s_wait_asynccnt)
#define HAVE_WAIT_ASYNC_BUILTIN 1
#endif
#endif

#if defined(ASYNC_LDS)
// Builtin signature (from the round-2 diagnostic): param1 is
// 'int __attribute__((vector_size(16))) __device__ *' (v4i32, AS1, non-const).
typedef int v4i_t __attribute__((vector_size(4 * sizeof(int))));
typedef __attribute__((address_space(1))) v4i_t* g_v4i_p;
typedef __attribute__((address_space(3))) v4i_t* l_v4i_p;
// generic->AS1: same 64-bit address; generic LDS ptr low 32 bits == LDS offset
__device__ __forceinline__ g_v4i_p as_global(const void* p) {
  return (g_v4i_p)(uintptr_t)p;
}
__device__ __forceinline__ l_v4i_p as_lds(void* p) {
  return (l_v4i_p)(unsigned)(uintptr_t)p;
}
__device__ __forceinline__ void wait_async0() {
#if defined(HAVE_WAIT_ASYNC_BUILTIN)
  __builtin_amdgcn_s_wait_asynccnt(0);
#else
  asm volatile("s_wait_asynccnt 0x0" ::: "memory");
#endif
}
#endif

// --- WMMA fragment loaders (layouts per CDNA5 ISA 7.12.2) -------------------
// A-matrix 16x32 bf16: lanes 0-15 (row = lane) hold K = 0..7 (V0-3), 16..23
// (V4-7); lanes 16-31 hold K = 8..15, 24..31.  => lo @ half*8, hi @ half*8+16.
__device__ __forceinline__ bf16x16 frag_a(const __bf16* rowp, int half) {
  bf16x8 lo = *(const bf16x8*)(rowp + half * 8);
  bf16x8 hi = *(const bf16x8*)(rowp + half * 8 + 16);
  return __builtin_shufflevector(lo, hi, 0, 1, 2, 3, 4, 5, 6, 7,
                                 8, 9, 10, 11, 12, 13, 14, 15);
}
// B-matrix 32x16 bf16: lanes 0-15 (col = lane) hold K = 0..7, 16..23;
// lanes 16-31 hold K = 8..15, 24..31.  => lo @ half*8, hi @ half*8+16.
__device__ __forceinline__ bf16x16 frag_b(const __bf16* colp, int half) {
  bf16x8 lo = *(const bf16x8*)(colp + half * 8);
  bf16x8 hi = *(const bf16x8*)(colp + half * 8 + 16);
  return __builtin_shufflevector(lo, hi, 0, 1, 2, 3, 4, 5, 6, 7,
                                 8, 9, 10, 11, 12, 13, 14, 15);
}

__device__ __forceinline__ f32x8 wmma_bf16(bf16x16 a, bf16x16 b, f32x8 c) {
  return __builtin_amdgcn_wmma_f32_16x16x32_bf16(false, a, false, b,
                                                 (short)0, c, false, false);
}

// --- fp32 -> bf16 conversion ------------------------------------------------
__global__ __launch_bounds__(256) void cvt_f32_bf16(const float* __restrict__ in,
                                                    __bf16* __restrict__ out,
                                                    int n) {
  int i = (blockIdx.x * 256 + threadIdx.x) * 4;
  if (i + 4 <= n) {
    float4 v = *(const float4*)(in + i);
    out[i + 0] = (__bf16)v.x;
    out[i + 1] = (__bf16)v.y;
    out[i + 2] = (__bf16)v.z;
    out[i + 3] = (__bf16)v.w;
  } else {
    for (int j = i; j < n; ++j) out[j] = (__bf16)in[j];
  }
}

// --- Generic NT GEMM: C[M,N] = A[M,K] @ B[N,K]^T ---------------------------
// Block tile 128x128, 8 waves (4x2), each wave 32x64 = 2x4 16x16 accumulators,
// K-step 32, double-buffered LDS, pipelined global fetch.
// MODE 0: QKV scatter (q scaled 1/8, bf16). MODE 1: +bias,ReLU,bf16.
// MODE 2: +bias, fp32.
template <int MODE>
__global__ __launch_bounds__(256)
void gemm_wmma_nt(const __bf16* __restrict__ A, const __bf16* __restrict__ Bw,
                  const float* __restrict__ bias, float* __restrict__ outF,
                  __bf16* __restrict__ outB, __bf16* __restrict__ qOut,
                  __bf16* __restrict__ kOut, __bf16* __restrict__ vOut,
                  int M, int N, int K) {
  __shared__ __align__(16) __bf16 lA[2][128 * 40];  // stride 40: bank-safe b128
  __shared__ __align__(16) __bf16 lB[2][128 * 40];

  const int tid = threadIdx.x;
  const int lane = tid & 31, w = tid >> 5;
  const int half = lane >> 4, r15 = lane & 15;
  const int wm = w >> 1, wn = w & 1;  // 4 (M) x 2 (N) wave grid
  const int bm = blockIdx.y * 128, bn = blockIdx.x * 128;

  // per-thread tile-copy coordinates: 512 16B-chunks per matrix, 2 per thread
  const int r0 = tid >> 2, r1 = r0 + 64, ch = (tid & 3) << 3;

  const f32x8 z8 = {0.f, 0.f, 0.f, 0.f, 0.f, 0.f, 0.f, 0.f};
  f32x8 acc[2][4];
#pragma unroll
  for (int i = 0; i < 2; ++i)
#pragma unroll
    for (int j = 0; j < 4; ++j) acc[i][j] = z8;

  const int nkb = K >> 5;

#if defined(ASYNC_LDS)
  // ---- pipelined path: GLOBAL_LOAD_ASYNC_TO_LDS_B128 (ASYNCcnt) ----
  auto issue_tile = [&](int kb, int p) {
    const int k0 = kb << 5;
    __builtin_amdgcn_global_load_async_to_lds_b128(
        as_global(A + (size_t)(bm + r0) * K + k0 + ch),
        as_lds(&lA[p][r0 * 40 + ch]), 0, 0);
    __builtin_amdgcn_global_load_async_to_lds_b128(
        as_global(A + (size_t)(bm + r1) * K + k0 + ch),
        as_lds(&lA[p][r1 * 40 + ch]), 0, 0);
    __builtin_amdgcn_global_load_async_to_lds_b128(
        as_global(Bw + (size_t)(bn + r0) * K + k0 + ch),
        as_lds(&lB[p][r0 * 40 + ch]), 0, 0);
    __builtin_amdgcn_global_load_async_to_lds_b128(
        as_global(Bw + (size_t)(bn + r1) * K + k0 + ch),
        as_lds(&lB[p][r1 * 40 + ch]), 0, 0);
  };
  issue_tile(0, 0);
  for (int kb = 0; kb < nkb; ++kb) {
    const int p = kb & 1;
    wait_async0();      // tile kb resident in LDS
    __syncthreads();    // all waves see it / all done reading buf p^1
    if (kb + 1 < nkb) issue_tile(kb + 1, p ^ 1);  // overlaps with WMMAs below
    bf16x16 bfr[4];
#pragma unroll
    for (int ni = 0; ni < 4; ++ni)
      bfr[ni] = frag_b(&lB[p][(wn * 64 + ni * 16 + r15) * 40], half);
#pragma unroll
    for (int mi = 0; mi < 2; ++mi) {
      bf16x16 afr = frag_a(&lA[p][(wm * 32 + mi * 16 + r15) * 40], half);
#pragma unroll
      for (int ni = 0; ni < 4; ++ni)
        acc[mi][ni] = wmma_bf16(afr, bfr[ni], acc[mi][ni]);
    }
  }
#else
  // ---- fallback: register-staged double buffer ----
  bf16x8 tA0, tA1, tB0, tB1;
  auto fetch_tile = [&](int kb) {
    const int k0 = kb << 5;
    tA0 = *(const bf16x8*)(A + (size_t)(bm + r0) * K + k0 + ch);
    tA1 = *(const bf16x8*)(A + (size_t)(bm + r1) * K + k0 + ch);
    tB0 = *(const bf16x8*)(Bw + (size_t)(bn + r0) * K + k0 + ch);
    tB1 = *(const bf16x8*)(Bw + (size_t)(bn + r1) * K + k0 + ch);
  };
  auto commit_tile = [&](int p) {
    *(bf16x8*)&lA[p][r0 * 40 + ch] = tA0;
    *(bf16x8*)&lA[p][r1 * 40 + ch] = tA1;
    *(bf16x8*)&lB[p][r0 * 40 + ch] = tB0;
    *(bf16x8*)&lB[p][r1 * 40 + ch] = tB1;
  };
  fetch_tile(0);
  commit_tile(0);
  for (int kb = 0; kb < nkb; ++kb) {
    const int p = kb & 1;
    __syncthreads();  // compiler-inserted dscnt wait makes tile kb visible
    if (kb + 1 < nkb) fetch_tile(kb + 1);  // global latency hidden by WMMAs
    bf16x16 bfr[4];
#pragma unroll
    for (int ni = 0; ni < 4; ++ni)
      bfr[ni] = frag_b(&lB[p][(wn * 64 + ni * 16 + r15) * 40], half);
#pragma unroll
    for (int mi = 0; mi < 2; ++mi) {
      bf16x16 afr = frag_a(&lA[p][(wm * 32 + mi * 16 + r15) * 40], half);
#pragma unroll
      for (int ni = 0; ni < 4; ++ni)
        acc[mi][ni] = wmma_bf16(afr, bfr[ni], acc[mi][ni]);
    }
    if (kb + 1 < nkb) commit_tile(p ^ 1);
  }
#endif

  // Epilogue. C-frag: lane = col (r15), VGPR v = row (v + 8*half).
#pragma unroll
  for (int mi = 0; mi < 2; ++mi) {
    int gm0 = bm + wm * 32 + mi * 16 + half * 8;
#pragma unroll
    for (int ni = 0; ni < 4; ++ni) {
      int gn = bn + wn * 64 + ni * 16 + r15;
      if (MODE == 0) {
        int hh = gn / 192, c = gn % 192;
        __bf16* dst = (c < 64) ? qOut : (c < 128 ? kOut : vOut);
        int dc = c & 63;
        float scale = (c < 64) ? 0.125f : 1.0f;  // 1/sqrt(64) folded into q
#pragma unroll
        for (int v = 0; v < 8; ++v) {
          int gm = gm0 + v;
          int bidx = gm >> 11, nrow = gm & 2047;
          dst[(((size_t)bidx * NHEAD + hh) * NSEQ + nrow) * DHEAD + dc] =
              (__bf16)(acc[mi][ni][v] * scale);
        }
      } else if (MODE == 1) {
        float bb = bias[gn];
#pragma unroll
        for (int v = 0; v < 8; ++v) {
          float x = acc[mi][ni][v] + bb;
          outB[(size_t)(gm0 + v) * N + gn] = (__bf16)fmaxf(x, 0.f);
        }
      } else {
        float bb = bias[gn];
#pragma unroll
        for (int v = 0; v < 8; ++v)
          outF[(size_t)(gm0 + v) * N + gn] = acc[mi][ni][v] + bb;
      }
    }
  }
}

// --- Flash attention (per b,h): wave = 16 query rows, 32-key column blocks --
__global__ __launch_bounds__(256)
void attn_kernel(const __bf16* __restrict__ Q, const __bf16* __restrict__ Kb,
                 const __bf16* __restrict__ V, float* __restrict__ out) {
  __shared__ __align__(16) __bf16 sK[32 * 72];      // [m][d], stride 72
  __shared__ __align__(16) __bf16 sVt[64 * 40];     // [d][m], stride 40
  __shared__ float sS[8 * 16 * 33];                 // per-wave scores
  __shared__ __align__(16) __bf16 sP[8 * 16 * 40];  // per-wave probs (A-frag)
  __shared__ float sStat[8 * 16];                   // alpha / 1/l

  const int tid = threadIdx.x, lane = tid & 31, w = tid >> 5;
  const int half = lane >> 4, r15 = lane & 15;
  const int bh = blockIdx.y;
  const int q0 = blockIdx.x * 128 + w * 16;
  const size_t base = (size_t)bh * NSEQ * DHEAD;

  // Q A-frags, K-dim = d (two 32-chunks). q already scaled by 1/sqrt(D).
  bf16x16 aq[2];
#pragma unroll
  for (int kc = 0; kc < 2; ++kc)
    aq[kc] = frag_a(Q + base + (size_t)(q0 + r15) * DHEAD + kc * 32, half);

  const f32x8 z8 = {0.f, 0.f, 0.f, 0.f, 0.f, 0.f, 0.f, 0.f};
  f32x8 oacc[4];
#pragma unroll
  for (int ni = 0; ni < 4; ++ni) oacc[ni] = z8;
  float m_i = -1e30f, l_i = 0.f;

  for (int cb = 0; cb < NSEQ / 32; ++cb) {
    const int m0 = cb * 32;
    {  // stage K tile [32][64] and V^T tile [64][32]; 256 chunks = 1/thread
      int mrow = tid >> 3, ch = (tid & 7) << 3;
      *(bf16x8*)&sK[mrow * 72 + ch] =
          *(const bf16x8*)(Kb + base + (size_t)(m0 + mrow) * DHEAD + ch);
      bf16x8 vv = *(const bf16x8*)(V + base + (size_t)(m0 + mrow) * DHEAD + ch);
#pragma unroll
      for (int i = 0; i < 8; ++i) sVt[(ch + i) * 40 + mrow] = vv[i];
      if (cb + 1 < NSEQ / 32) {  // global_prefetch_b8 for the next tile
        __builtin_prefetch(Kb + base + (size_t)(m0 + 32 + mrow) * DHEAD + ch, 0,
                           1);
        __builtin_prefetch(V + base + (size_t)(m0 + 32 + mrow) * DHEAD + ch, 0,
                           1);
      }
    }
    __syncthreads();

    // S = Q @ K^T : 2 n-tiles x 2 K-chunks
    f32x8 sacc[2];
    sacc[0] = z8;
    sacc[1] = z8;
#pragma unroll
    for (int nt = 0; nt < 2; ++nt) {
      const __bf16* kr = &sK[(nt * 16 + r15) * 72];
#pragma unroll
      for (int kc = 0; kc < 2; ++kc)
        sacc[nt] = wmma_bf16(aq[kc], frag_b(kr + kc * 32, half), sacc[nt]);
    }
#pragma unroll
    for (int nt = 0; nt < 2; ++nt)
#pragma unroll
      for (int v = 0; v < 8; ++v)
        sS[(w * 16 + v + half * 8) * 33 + nt * 16 + r15] = sacc[nt][v];
    __syncthreads();

    // Online softmax: lanes 0-15 each own one row of their wave's tile.
    if (half == 0) {
      const float* srow = &sS[(w * 16 + r15) * 33];
      float mn = m_i;
#pragma unroll
      for (int j = 0; j < 32; ++j) mn = fmaxf(mn, srow[j]);
      float alpha = __expf(m_i - mn);
      float ls = 0.f;
      __bf16* prow = &sP[(w * 16 + r15) * 40];
#pragma unroll
      for (int j = 0; j < 32; ++j) {
        float p = __expf(srow[j] - mn);
        ls += p;
        prow[j] = (__bf16)p;
      }
      l_i = l_i * alpha + ls;
      m_i = mn;
      sStat[w * 16 + r15] = alpha;
    }
    __syncthreads();

    // Rescale O, then O += P @ V (K-dim = 32 keys -> one WMMA per d-tile).
    float al[8];
#pragma unroll
    for (int v = 0; v < 8; ++v) al[v] = sStat[w * 16 + v + half * 8];
#pragma unroll
    for (int ni = 0; ni < 4; ++ni)
#pragma unroll
      for (int v = 0; v < 8; ++v) oacc[ni][v] *= al[v];
    bf16x16 ap = frag_a(&sP[(w * 16 + r15) * 40], half);
#pragma unroll
    for (int ni = 0; ni < 4; ++ni)
      oacc[ni] = wmma_bf16(ap, frag_b(&sVt[(ni * 16 + r15) * 40], half),
                           oacc[ni]);
    __syncthreads();
  }

  if (half == 0) sStat[w * 16 + r15] = 1.f / l_i;
  __syncthreads();
  float il[8];
#pragma unroll
  for (int v = 0; v < 8; ++v) il[v] = sStat[w * 16 + v + half * 8];
  const int b = bh >> 4, hh = bh & 15;
#pragma unroll
  for (int ni = 0; ni < 4; ++ni)
#pragma unroll
    for (int v = 0; v < 8; ++v)
      out[((size_t)b * NSEQ + q0 + v + half * 8) * DMODEL + hh * DHEAD +
          ni * 16 + r15] = oacc[ni][v] * il[v];
}

// --- residual + LayerNorm (one block per token row) ------------------------
__global__ __launch_bounds__(256)
void add_ln_kernel(const float* __restrict__ X, const float* __restrict__ R,
                   const float* __restrict__ G, const float* __restrict__ Bb,
                   float* __restrict__ Yf, __bf16* __restrict__ Yb) {
  __shared__ float red[256];
  const int row = blockIdx.x, tid = threadIdx.x;
  const float* xr = X + (size_t)row * DMODEL;
  const float* rr = R + (size_t)row * DMODEL;
  float loc[4];
  float s = 0.f;
#pragma unroll
  for (int i = 0; i < 4; ++i) {
    int idx = tid + i * 256;
    loc[i] = xr[idx] + rr[idx];
    s += loc[i];
  }
  red[tid] = s;
  __syncthreads();
  for (int o = 128; o > 0; o >>= 1) {
    if (tid < o) red[tid] += red[tid + o];
    __syncthreads();
  }
  float mu = red[0] * (1.f / DMODEL);
  __syncthreads();
  float s2 = 0.f;
#pragma unroll
  for (int i = 0; i < 4; ++i) {
    float d = loc[i] - mu;
    s2 += d * d;
  }
  red[tid] = s2;
  __syncthreads();
  for (int o = 128; o > 0; o >>= 1) {
    if (tid < o) red[tid] += red[tid + o];
    __syncthreads();
  }
  float rstd = rsqrtf(red[0] * (1.f / DMODEL) + 1e-5f);
#pragma unroll
  for (int i = 0; i < 4; ++i) {
    int idx = tid + i * 256;
    float y = (loc[i] - mu) * rstd * G[idx] + Bb[idx];
    Yf[(size_t)row * DMODEL + idx] = y;
    if (Yb) Yb[(size_t)row * DMODEL + idx] = (__bf16)y;
  }
}

// ---------------------------------------------------------------------------
extern "C" void kernel_launch(void* const* d_in, const int* in_sizes, int n_in,
                              void* d_out, int out_size, void* d_ws,
                              size_t ws_size, hipStream_t stream) {
  const float* x    = (const float*)d_in[0];
  const float* Wqkv = (const float*)d_in[1];
  const float* ln1g = (const float*)d_in[2];
  const float* ln1b = (const float*)d_in[3];
  const float* W1   = (const float*)d_in[4];
  const float* b1   = (const float*)d_in[5];
  const float* W2   = (const float*)d_in[6];
  const float* b2   = (const float*)d_in[7];
  const float* ln2g = (const float*)d_in[8];
  const float* ln2b = (const float*)d_in[9];
  float* out = (float*)d_out;
  (void)in_sizes; (void)n_in; (void)out_size; (void)ws_size;

  char* ws = (char*)d_ws;
  size_t off = 0;
  auto take = [&](size_t bytes) -> char* {
    char* p = ws + off;
    off = (off + bytes + 255) & ~(size_t)255;
    return p;
  };
  const size_t MTOK = (size_t)BATCH * NSEQ;  // 8192 token rows
  __bf16* xb    = (__bf16*)take(MTOK * DMODEL * 2);
  __bf16* wqkvb = (__bf16*)take((size_t)3 * DMODEL * DMODEL * 2);
  __bf16* w1b   = (__bf16*)take((size_t)DMODEL * DMODEL * 2);
  __bf16* w2b   = (__bf16*)take((size_t)DMODEL * DMODEL * 2);
  __bf16* qb    = (__bf16*)take(MTOK * DMODEL * 2);
  __bf16* kb    = (__bf16*)take(MTOK * DMODEL * 2);
  __bf16* vb    = (__bf16*)take(MTOK * DMODEL * 2);
  float*  attn  = (float*)take(MTOK * DMODEL * 4);
  float*  yf    = (float*)take(MTOK * DMODEL * 4);
  __bf16* ybf   = (__bf16*)take(MTOK * DMODEL * 2);
  __bf16* h1    = (__bf16*)take(MTOK * DMODEL * 2);
  float*  z     = (float*)take(MTOK * DMODEL * 4);

  auto cvt = [&](const float* in, __bf16* o, size_t n) {
    cvt_f32_bf16<<<(unsigned)((n + 1023) / 1024), 256, 0, stream>>>(in, o,
                                                                    (int)n);
  };
  cvt(x, xb, MTOK * DMODEL);
  cvt(Wqkv, wqkvb, (size_t)3 * DMODEL * DMODEL);
  cvt(W1, w1b, (size_t)DMODEL * DMODEL);
  cvt(W2, w2b, (size_t)DMODEL * DMODEL);

  // qkv = x @ Wqkv^T, scattered to q/k/v [B,H,N,D] bf16 (q pre-scaled)
  gemm_wmma_nt<0><<<dim3(3 * DMODEL / 128, (unsigned)(MTOK / 128)), 256, 0,
                    stream>>>(xb, wqkvb, nullptr, nullptr, nullptr, qb, kb, vb,
                              (int)MTOK, 3 * DMODEL, DMODEL);
  // attention -> fp32 [B,N,DM]
  attn_kernel<<<dim3(NSEQ / 128, BATCH * NHEAD), 256, 0, stream>>>(qb, kb, vb,
                                                                   attn);
  // y = LN(x + attn)
  add_ln_kernel<<<(unsigned)MTOK, 256, 0, stream>>>(x, attn, ln1g, ln1b, yf,
                                                    ybf);
  // h = relu(y @ W1^T + b1) (bf16)
  gemm_wmma_nt<1><<<dim3(DMODEL / 128, (unsigned)(MTOK / 128)), 256, 0,
                    stream>>>(ybf, w1b, b1, nullptr, h1, nullptr, nullptr,
                              nullptr, (int)MTOK, DMODEL, DMODEL);
  // z = h @ W2^T + b2 (fp32)
  gemm_wmma_nt<2><<<dim3(DMODEL / 128, (unsigned)(MTOK / 128)), 256, 0,
                    stream>>>(h1, w2b, b2, z, nullptr, nullptr, nullptr,
                              nullptr, (int)MTOK, DMODEL, DMODEL);
  // out = LN(y + z)
  add_ln_kernel<<<(unsigned)MTOK, 256, 0, stream>>>(yf, z, ln2g, ln2b, out,
                                                    nullptr);
}
